// MultiHeadAttention_86028194939171
// MI455X (gfx1250) — matrix operand order repaired and checked by reading, not compiled, not run on hardware
//
#include <hip/hip_runtime.h>

// ---------------------------------------------------------------------------
// MI455X (gfx1250) fused multi-head attention, bf16 WMMA + fp32 accumulate.
//   K1: qkv = x @ Wqkv   (WMMA GEMM, double-buffered LDS, scatter q/k/v bf16;
//                         q pre-scaled by 1/sqrt(64))
//   K2: flash attention  (QK^T -> online softmax -> PV; K tile staged with
//                         CDNA5 async global->LDS, V double-buffered via regs)
//   K3: out = ctx @ Wo   (WMMA GEMM, fp32 out)
// B=2 T=2048 D=1024 H=16 DH=64
// ---------------------------------------------------------------------------

typedef __attribute__((ext_vector_type(16))) __bf16 v16bf;
typedef __attribute__((ext_vector_type(8)))  float  v8f;

__device__ __forceinline__ v8f v8f_zero() {
    v8f z = {0.f, 0.f, 0.f, 0.f, 0.f, 0.f, 0.f, 0.f};
    return z;
}

__device__ __forceinline__ v8f wmma_bf16(v16bf a, v16bf b, v8f c) {
    // (neg_a, A, neg_b, B, c_mod, C, reuse_a, reuse_b)
    return __builtin_amdgcn_wmma_f32_16x16x32_bf16(false, a, false, b,
                                                   (short)0, c, false, false);
}

// K element base for 16-bit A-fragment VGPR v, lane-half hi (contiguous, matches B):
// lanes 0-15: VGPR0..7 -> K 0..15; lanes 16-31: K 16..31
__device__ __forceinline__ int a_kbase(int v, int hi) {
    return (hi << 4) + (v << 1);
}

__device__ __forceinline__ unsigned pk_bf16(float lo, float hi) {
    union { __bf16 b[2]; unsigned u; } u_;
    u_.b[0] = (__bf16)lo;
    u_.b[1] = (__bf16)hi;
    return u_.u;
}

// XOR-butterfly via ds_swizzle (group-of-32: and=0x1F, or=0, xor=imm[14:10]).
// Cheaper than ds_bpermute: immediate pattern, no address VGPR, no NaN quiet.
template <int IMM>
__device__ __forceinline__ float swz_f32(float x) {
    return __builtin_bit_cast(
        float, __builtin_amdgcn_ds_swizzle(__builtin_bit_cast(int, x), IMM));
}
__device__ __forceinline__ float bfly_max16(float x) {
    x = fmaxf(x, swz_f32<0x041F>(x));   // xor 1
    x = fmaxf(x, swz_f32<0x081F>(x));   // xor 2
    x = fmaxf(x, swz_f32<0x101F>(x));   // xor 4
    x = fmaxf(x, swz_f32<0x201F>(x));   // xor 8  (stays inside 16-lane half)
    return x;
}
__device__ __forceinline__ float bfly_sum16(float x) {
    x += swz_f32<0x041F>(x);
    x += swz_f32<0x081F>(x);
    x += swz_f32<0x101F>(x);
    x += swz_f32<0x201F>(x);
    return x;
}

// gfx1250: generic pointers to LDS carry the wave-relative LDS byte offset in
// their low 32 bits (flat->LDS mapping truncates to addr[31:0]).
__device__ __forceinline__ unsigned lds_addr32(const void* p) {
    return (unsigned)(size_t)p;
}

// ---------------------------------------------------------------------------
// 128x128x32-tile GEMM: C[M,N] = A[M,K](f32) * B[K,N](f32), bf16 WMMA,
// double-buffered LDS.  MODE 0: fp32 C.  MODE 1: scatter bf16 q/k/v
// (head layout [B,H,T,64]), q scaled by 0.125 = 1/sqrt(DH).
// 256 threads = 8 waves as 4(M) x 2(N); each wave: 32x64 of C.
// ---------------------------------------------------------------------------
template <int MODE>
__global__ __launch_bounds__(256)
void gemm_bf16_wmma(const float* __restrict__ A, const float* __restrict__ B,
                    int K, int N, float* __restrict__ C,
                    __bf16* __restrict__ qws, __bf16* __restrict__ kws,
                    __bf16* __restrict__ vws)
{
    __shared__ __align__(16) __bf16 As[2][128][40];   // [m][k], +8 pad
    __shared__ __align__(16) __bf16 Bs[2][128][40];   // [n][k] transposed

    const int tid  = threadIdx.x;
    const int lane = tid & 31;
    const int wv   = tid >> 5;
    const int wm   = wv & 3;
    const int wn   = wv >> 2;
    const int hi   = lane >> 4;
    const int ln   = lane & 15;

    const int blockM = blockIdx.y * 128;
    const int blockN = blockIdx.x * 128;

    const float* Abase = A + (size_t)blockM * K;
    const float* Bbase = B + blockN;

    v8f acc[2][4];
    #pragma unroll
    for (int i = 0; i < 2; ++i)
        #pragma unroll
        for (int j = 0; j < 4; ++j) acc[i][j] = v8f_zero();

    float4 areg[4], breg[4];

    auto load_tile = [&](int kt) {
        #pragma unroll
        for (int i = 0; i < 4; ++i) {
            int idx = tid * 4 + i;
            int r = idx >> 3, c = (idx & 7) << 2;       // A: 128 rows x 32
            areg[i] = *(const float4*)(Abase + (size_t)r * K + kt * 32 + c);
            int kk = idx >> 5, n = (idx & 31) << 2;     // B: 32 rows x 128
            breg[i] = *(const float4*)(Bbase + (size_t)(kt * 32 + kk) * N + n);
        }
    };
    auto store_tile = [&](int buf) {
        #pragma unroll
        for (int i = 0; i < 4; ++i) {
            int idx = tid * 4 + i;
            int r = idx >> 3, c = (idx & 7) << 2;
            *(unsigned*)&As[buf][r][c + 0] = pk_bf16(areg[i].x, areg[i].y);
            *(unsigned*)&As[buf][r][c + 2] = pk_bf16(areg[i].z, areg[i].w);
            int kk = idx >> 5, n = (idx & 31) << 2;     // transposed: no pack
            Bs[buf][n + 0][kk] = (__bf16)breg[i].x;
            Bs[buf][n + 1][kk] = (__bf16)breg[i].y;
            Bs[buf][n + 2][kk] = (__bf16)breg[i].z;
            Bs[buf][n + 3][kk] = (__bf16)breg[i].w;
        }
    };

    const int kTiles = K >> 5;
    load_tile(0);
    store_tile(0);
    __syncthreads();

    for (int kt = 0; kt < kTiles; ++kt) {
        const int cur  = kt & 1;
        const bool more = (kt + 1) < kTiles;
        if (more) load_tile(kt + 1);                   // global->regs, overlaps WMMA
        if (kt + 2 < kTiles) {                         // L2 prefetch 2 tiles ahead
            int idx = tid * 4;
            int r = idx >> 3, c = (idx & 7) << 2;
            __builtin_prefetch(Abase + (size_t)r * K + (kt + 2) * 32 + c, 0, 0);
            int kk = idx >> 5, n = (idx & 31) << 2;
            __builtin_prefetch(Bbase + (size_t)((kt + 2) * 32 + kk) * N + n, 0, 0);
        }

        v16bf af[2];
        #pragma unroll
        for (int tm = 0; tm < 2; ++tm) {
            int row = wm * 32 + tm * 16 + ln;
            unsigned* ap = (unsigned*)&af[tm];
            #pragma unroll
            for (int v = 0; v < 8; ++v)
                ap[v] = *(const unsigned*)&As[cur][row][a_kbase(v, hi)];
        }
        v16bf bfr[4];
        #pragma unroll
        for (int tn = 0; tn < 4; ++tn) {
            int col = wn * 64 + tn * 16 + ln;
            unsigned* bp = (unsigned*)&bfr[tn];
            #pragma unroll
            for (int v = 0; v < 8; ++v)
                bp[v] = *(const unsigned*)&Bs[cur][col][(hi << 4) + (v << 1)];
        }
        #pragma unroll
        for (int tm = 0; tm < 2; ++tm)
            #pragma unroll
            for (int tn = 0; tn < 4; ++tn)
                acc[tm][tn] = wmma_bf16(af[tm], bfr[tn], acc[tm][tn]);

        if (more) store_tile(1 - cur);
        __syncthreads();
    }

    // Epilogue. C layout: M = v + 8*hi, N = lane&15 per 16x16 tile.
    if (MODE == 0) {
        #pragma unroll
        for (int tm = 0; tm < 2; ++tm)
            #pragma unroll
            for (int tn = 0; tn < 4; ++tn)
                #pragma unroll
                for (int v = 0; v < 8; ++v) {
                    int m = blockM + wm * 32 + tm * 16 + v + (hi << 3);
                    int n = blockN + wn * 64 + tn * 16 + ln;
                    C[(size_t)m * N + n] = acc[tm][tn][v];
                }
    } else {
        #pragma unroll
        for (int tm = 0; tm < 2; ++tm) {
            #pragma unroll
            for (int tn = 0; tn < 4; ++tn) {
                // All uniform per 16-wide tile (never straddles a 64/1024
                // boundary) -> scalar selects, no divergent branches.
                int ncol  = blockN + wn * 64 + tn * 16;
                int which = ncol >> 10;                  // 0=q 1=k 2=v
                int d     = ncol & 1023;
                int h     = d >> 6;
                int dh    = (d & 63) + ln;               // per-lane
                __bf16* dst = (which == 0) ? qws : (which == 1) ? kws : vws;
                float   sc  = (which == 0) ? 0.125f : 1.0f;  // fold 1/sqrt(64)
                #pragma unroll
                for (int v = 0; v < 8; ++v) {
                    int m = blockM + wm * 32 + tm * 16 + v + (hi << 3);
                    int b = m >> 11, t = m & 2047;
                    dst[(((size_t)(b * 16 + h)) * 2048 + t) * 64 + dh] =
                        (__bf16)(acc[tm][tn][v] * sc);
                }
            }
        }
    }
}

// ---------------------------------------------------------------------------
// Flash attention: block = (b,h, 128 q rows), 8 waves x 16 q rows each.
// Double-buffered 64-key K/V blocks in LDS. K is a straight bf16 copy ->
// CDNA5 async global->LDS (ASYNCcnt); V needs a transpose -> via registers.
// Per block: S = Q K^T (2 WMMA/tile), online softmax, O += P V.
// ---------------------------------------------------------------------------
__global__ __launch_bounds__(256)
void flash_attn_wmma(const __bf16* __restrict__ qws, const __bf16* __restrict__ kws,
                     const __bf16* __restrict__ vws, float* __restrict__ ctx)
{
    __shared__ __align__(16) __bf16 Ks[2][64][72];     // [key][dh]  (+8 pad)
    __shared__ __align__(16) __bf16 Vs[2][64][72];     // [dh][key]  transposed
    __shared__ __align__(16) __bf16 Pw[8][16][72];     // per-wave P strip

    const int tid  = threadIdx.x;
    const int lane = tid & 31;
    const int wv   = tid >> 5;
    const int hi   = lane >> 4;
    const int ln   = lane & 15;

    const int bh = blockIdx.y;                 // 0..31
    const int b  = bh >> 4;
    const int h  = bh & 15;
    const int q0 = blockIdx.x * 128 + wv * 16; // wave's first q row

    const __bf16* qb  = qws + (size_t)bh * 2048 * 64;
    const __bf16* kbp = kws + (size_t)bh * 2048 * 64;
    const __bf16* vbp = vws + (size_t)bh * 2048 * 64;

    // Q strip as two 16x32 A-fragments (resident; already scaled by 1/sqrt(64))
    v16bf qa[2];
    {
        const __bf16* qrow = qb + (size_t)(q0 + ln) * 64;
        #pragma unroll
        for (int f = 0; f < 2; ++f) {
            unsigned* p = (unsigned*)&qa[f];
            #pragma unroll
            for (int v = 0; v < 8; ++v)
                p[v] = *(const unsigned*)(qrow + f * 32 + a_kbase(v, hi));
        }
    }

    v8f o[4];
    #pragma unroll
    for (int dt = 0; dt < 4; ++dt) o[dt] = v8f_zero();
    float mrow[8], lrow[8];
    #pragma unroll
    for (int v = 0; v < 8; ++v) { mrow[v] = -3.0e38f; lrow[v] = 0.f; }

    // --- K tile: async global->LDS, 512 x B128 across the block -----------
    auto async_load_k = [&](int kb, int buf) {
        #pragma unroll
        for (int i = 0; i < 2; ++i) {
            int idx = tid * 2 + i;
            int r = idx >> 3, c = (idx & 7) << 3;
            unsigned      lds = lds_addr32(&Ks[buf][r][c]);
            const __bf16* g   = kbp + (size_t)kb * 4096 + r * 64 + c;
            asm volatile("global_load_async_to_lds_b128 %0, %1, off"
                         :: "v"(lds), "v"(g) : "memory");
        }
    };
    // --- V tile: global->regs->LDS (transposed) ----------------------------
    unsigned vreg[8];
    auto load_v = [&](int kb) {
        #pragma unroll
        for (int i = 0; i < 8; ++i) {
            int idx = tid * 8 + i;             // 2048 x uint (2 bf16)
            int key = idx >> 5, dh2 = (idx & 31) << 1;
            vreg[i] = *(const unsigned*)(vbp + (size_t)kb * 4096 + key * 64 + dh2);
        }
    };
    auto store_v = [&](int buf) {
        #pragma unroll
        for (int i = 0; i < 8; ++i) {
            int idx = tid * 8 + i;
            int key = idx >> 5, dh2 = (idx & 31) << 1;
            Vs[buf][dh2 + 0][key] = ((const __bf16*)&vreg[i])[0];
            Vs[buf][dh2 + 1][key] = ((const __bf16*)&vreg[i])[1];
        }
    };

    async_load_k(0, 0);
    load_v(0);
    store_v(0);
    asm volatile("s_wait_asynccnt 0" ::: "memory");
    __syncthreads();

    for (int kb = 0; kb < 32; ++kb) {          // 2048 keys / 64
        const int  cur  = kb & 1;
        const bool more = (kb + 1) < 32;
        if (more) {
            async_load_k(kb + 1, 1 - cur);     // async K copy overlaps WMMA
            load_v(kb + 1);                    // global->regs overlaps WMMA
        }
        if (kb + 2 < 32) {
            __builtin_prefetch(kbp + (size_t)(kb + 2) * 4096 + tid * 16, 0, 0);
            __builtin_prefetch(vbp + (size_t)(kb + 2) * 4096 + tid * 16, 0, 0);
        }

        // S strip: 4 tiles of 16x16, K-dim = 64 (2 WMMAs each)
        v8f s[4];
        #pragma unroll
        for (int nt = 0; nt < 4; ++nt) {
            v16bf b0, b1;
            unsigned* p0 = (unsigned*)&b0;
            unsigned* p1 = (unsigned*)&b1;
            int key = nt * 16 + ln;            // B column = key index
            #pragma unroll
            for (int v = 0; v < 8; ++v) {
                int kd = (hi << 4) + (v << 1); // B K-pairs (dh)
                p0[v] = *(const unsigned*)&Ks[cur][key][kd];
                p1[v] = *(const unsigned*)&Ks[cur][key][kd + 32];
            }
            v8f z = v8f_zero();
            z     = wmma_bf16(qa[0], b0, z);
            s[nt] = wmma_bf16(qa[1], b1, z);
        }

        // Online softmax per q row (row = v + 8*hi; its 16 N vals live in one
        // 16-lane half, so xor-1/2/4/8 swizzles reduce each half separately).
        #pragma unroll
        for (int v = 0; v < 8; ++v) {
            float rmax = -3.0e38f;
            #pragma unroll
            for (int nt = 0; nt < 4; ++nt) rmax = fmaxf(rmax, s[nt][v]);
            rmax = bfly_max16(rmax);
            float mnew = fmaxf(mrow[v], rmax);
            float corr = __expf(mrow[v] - mnew);
            mrow[v] = mnew;
            float psum = 0.f;
            #pragma unroll
            for (int nt = 0; nt < 4; ++nt) {
                float p = __expf(s[nt][v] - mnew);
                s[nt][v] = p;
                psum += p;
            }
            psum = bfly_sum16(psum);
            lrow[v] = lrow[v] * corr + psum;
            #pragma unroll
            for (int dt = 0; dt < 4; ++dt) o[dt][v] *= corr;
        }

        // P: C-layout -> per-wave LDS -> A-fragment layout
        #pragma unroll
        for (int nt = 0; nt < 4; ++nt)
            #pragma unroll
            for (int v = 0; v < 8; ++v)
                Pw[wv][v + (hi << 3)][nt * 16 + ln] = (__bf16)s[nt][v];

        asm volatile("s_wait_dscnt 0" ::: "memory");  // wave-local LDS RAW

        v16bf pa[2];
        #pragma unroll
        for (int f = 0; f < 2; ++f) {
            unsigned* pp = (unsigned*)&pa[f];
            #pragma unroll
            for (int v = 0; v < 8; ++v)
                pp[v] = *(const unsigned*)&Pw[wv][ln][f * 32 + a_kbase(v, hi)];
        }

        // O += P V : 4 dh tiles, K-dim = 64 keys (2 WMMAs each)
        #pragma unroll
        for (int dt = 0; dt < 4; ++dt) {
            v16bf vb0, vb1;
            unsigned* p0 = (unsigned*)&vb0;
            unsigned* p1 = (unsigned*)&vb1;
            int dh = dt * 16 + ln;             // B column = dh
            #pragma unroll
            for (int v = 0; v < 8; ++v) {
                int kd = (hi << 4) + (v << 1); // B K-pairs (key)
                p0[v] = *(const unsigned*)&Vs[cur][dh][kd];
                p1[v] = *(const unsigned*)&Vs[cur][dh][kd + 32];
            }
            o[dt] = wmma_bf16(pa[0], vb0, o[dt]);
            o[dt] = wmma_bf16(pa[1], vb1, o[dt]);
        }

        if (more) {
            store_v(1 - cur);                  // fill back buffer (V)
            asm volatile("s_wait_asynccnt 0" ::: "memory");  // K landed in LDS
        }
        __syncthreads();
    }

    // Normalize and write context fp32 [B*T, 1024]
    #pragma unroll
    for (int v = 0; v < 8; ++v) {
        float inv = 1.f / lrow[v];
        int t = q0 + v + (hi << 3);
        float* dst = ctx + ((size_t)(b * 2048 + t)) * 1024 + h * 64;
        #pragma unroll
        for (int dt = 0; dt < 4; ++dt)
            dst[dt * 16 + ln] = o[dt][v] * inv;
    }
}

// ---------------------------------------------------------------------------
extern "C" void kernel_launch(void* const* d_in, const int* in_sizes, int n_in,
                              void* d_out, int out_size, void* d_ws, size_t ws_size,
                              hipStream_t stream) {
    const float* x    = (const float*)d_in[0];   // [2,2048,1024]
    const float* Wqkv = (const float*)d_in[1];   // [1024,3072]
    const float* Wo   = (const float*)d_in[2];   // [1024,1024]
    float* out = (float*)d_out;                  // [2,2048,1024]

    const size_t nqkv = (size_t)2 * 16 * 2048 * 64;  // 4,194,304 per tensor
    __bf16* qws = (__bf16*)d_ws;                     //  8 MB
    __bf16* kws = qws + nqkv;                        //  8 MB
    __bf16* vws = kws + nqkv;                        //  8 MB
    float*  ctx = (float*)(vws + nqkv);              // 16 MB  (total ~40 MB)

    // K1: qkv projection, M=4096 N=3072 K=1024
    gemm_bf16_wmma<1><<<dim3(3072 / 128, 4096 / 128), 256, 0, stream>>>(
        x, Wqkv, 1024, 3072, nullptr, qws, kws, vws);

    // K2: flash attention, grid = (q tiles of 128, B*H)
    flash_attn_wmma<<<dim3(2048 / 128, 32), 256, 0, stream>>>(qws, kws, vws, ctx);

    // K3: output projection, M=4096 N=1024 K=1024
    gemm_bf16_wmma<0><<<dim3(1024 / 128, 4096 / 128), 256, 0, stream>>>(
        ctx, Wo, 1024, 1024, out, nullptr, nullptr, nullptr);
}
